// GNN_Model_11965778886719
// MI455X (gfx1250) — matrix operand order repaired and checked by reading, not compiled, not run on hardware
//
#include <hip/hip_runtime.h>

typedef __attribute__((ext_vector_type(16))) _Float16 v16h;
typedef __attribute__((ext_vector_type(8)))  _Float16 v8h;
typedef __attribute__((ext_vector_type(8)))  float    v8f;

constexpr int B   = 16;
constexpr int N   = 768;
constexpr int FIN = 56;
constexpr int D   = 140;
constexpr int L   = 4;
constexpr int DF  = 128;
constexpr int Dp  = 160;   // D padded to multiple of 32 (WMMA K granularity)

// ---------------------------------------------------------------------------
// WMMA fragment loaders (CDNA5 wave32 16x16x32 f16; k order contiguous, k = (lane>>4)*16 + e)
// ---------------------------------------------------------------------------
// A 16x32 f16 (row-major source): lane m = lane&15, K-half = (lane>>4)*16;
// one contiguous 32B run per lane.
__device__ __forceinline__ v16h load_a_frag(const _Float16* __restrict__ base,
                                            int lda, int mbase, int kbase, int lane) {
    const _Float16* p = base + (size_t)(mbase + (lane & 15)) * lda
                             + kbase + ((lane >> 4) << 4);
    v8h lo = *(const v8h*)(p);
    v8h hi = *(const v8h*)(p + 8);
    v16h a;
#pragma unroll
    for (int t = 0; t < 8; ++t) { a[t] = lo[t]; a[t + 8] = hi[t]; }
    return a;
}

// B 32x16 f16 from an NT-layout source Bt[n][k] (ldb = k-extent):
// lane column n = nbase + (lane&15); K = kbase + (lane>>4)*16 + e
// -> one contiguous 32B read per lane.
__device__ __forceinline__ v16h load_bt_frag(const _Float16* __restrict__ base,
                                             int ldb, int nbase, int kbase, int lane) {
    const _Float16* p = base + (size_t)(nbase + (lane & 15)) * ldb
                             + kbase + ((lane >> 4) << 4);
    return *(const v16h*)(p);
}

__device__ __forceinline__ v8f wmma_f16(v16h a, v16h b, v8f c) {
    return __builtin_amdgcn_wmma_f32_16x16x32_f16(false, a, false, b,
                                                  (short)0, c, false, false);
}

// ---------------------------------------------------------------------------
// Setup / elementwise kernels
// ---------------------------------------------------------------------------
// x = c_hs @ W_emb, also emits padded f16 copy (fused, kills a conversion pass)
__global__ void k_embed(const float* __restrict__ hs, const float* __restrict__ We,
                        float* __restrict__ x, _Float16* __restrict__ x16) {
    int idx = blockIdx.x * blockDim.x + threadIdx.x;
    if (idx >= B * N * Dp) return;
    int d  = idx % Dp;
    int bn = idx / Dp;
    float s = 0.f;
    if (d < D) {
        const float* hr = hs + (size_t)bn * FIN;
#pragma unroll 8
        for (int f = 0; f < FIN; ++f) s = fmaf(hr[f], We[f * D + d], s);
        x[(size_t)bn * D + d] = s;
    }
    x16[idx] = (_Float16)s;
}

__global__ void k_rcount(const int* __restrict__ valid, int* __restrict__ rbuf) {
    __shared__ int red[256];
    int b = blockIdx.x, t = threadIdx.x;
    int s = 0;
    for (int i = t; i < N; i += 256) s += (valid[(size_t)b * N + i] > 0) ? 1 : 0;
    red[t] = s; __syncthreads();
    for (int k = 128; k; k >>= 1) { if (t < k) red[t] += red[t + k]; __syncthreads(); }
    if (t == 0) rbuf[b] = red[0];
}

// W / A packed transposed (Wt[n][k] = W[k][n]) for contiguous B-fragment reads
__global__ void k_convw(const float* __restrict__ W, const float* __restrict__ A,
                        _Float16* __restrict__ Wt, _Float16* __restrict__ At) {
    int idx = blockIdx.x * blockDim.x + threadIdx.x;
    if (idx >= Dp * Dp) return;
    int n = idx / Dp, k = idx % Dp;
    bool v = (n < D) && (k < D);
    Wt[idx] = v ? (_Float16)W[k * D + n] : (_Float16)0.f;
    At[idx] = v ? (_Float16)A[k * D + n] : (_Float16)0.f;
}

// ---------------------------------------------------------------------------
// WMMA GEMMs: one wave = 32x32 output (2x2 tiles, 4 accumulators)
// ---------------------------------------------------------------------------
// out16[b,N,Dp] = A16[b] (NxDp) @ Wt^T (+bias on cols<D)
// optionally also writes outT[Dp][N] via contiguous v8h stores.
__global__ __launch_bounds__(32)
void k_gemm_f16(const _Float16* __restrict__ A, const _Float16* __restrict__ Wt,
                const float* __restrict__ bias, _Float16* __restrict__ out,
                _Float16* __restrict__ outT) {
    int lane = threadIdx.x;
    int mb = blockIdx.x << 5, nb = blockIdx.y << 5, b = blockIdx.z;
    const _Float16* Ab = A + (size_t)b * N * Dp;
    v8f acc[2][2] = {};
#pragma unroll
    for (int kb = 0; kb < Dp; kb += 32) {
        v16h a0 = load_a_frag(Ab, Dp, mb,      kb, lane);
        v16h a1 = load_a_frag(Ab, Dp, mb + 16, kb, lane);
        v16h b0 = load_bt_frag(Wt, Dp, nb,      kb, lane);
        v16h b1 = load_bt_frag(Wt, Dp, nb + 16, kb, lane);
        acc[0][0] = wmma_f16(a0, b0, acc[0][0]);
        acc[0][1] = wmma_f16(a0, b1, acc[0][1]);
        acc[1][0] = wmma_f16(a1, b0, acc[1][0]);
        acc[1][1] = wmma_f16(a1, b1, acc[1][1]);
    }
    int half = lane >> 4;
    _Float16* ob  = out  + (size_t)b * N * Dp;
    _Float16* obT = outT ? outT + (size_t)b * Dp * N : nullptr;
#pragma unroll
    for (int i = 0; i < 2; ++i) {
#pragma unroll
        for (int j = 0; j < 2; ++j) {
            int col = nb + j * 16 + (lane & 15);
            float bv = (bias != nullptr && col < D) ? bias[col] : 0.f;
            int rowb = mb + i * 16 + half * 8;
            v8h pack;
#pragma unroll
            for (int r = 0; r < 8; ++r) {
                float v = acc[i][j][r] + bv;
                pack[r] = (_Float16)v;
                ob[(size_t)(rowb + r) * Dp + col] = (_Float16)v;
            }
            if (obT) *(v8h*)&obT[(size_t)col * N + rowb] = pack;  // contiguous 16B
        }
    }
}

// e[b,N,N] = hA16[b] @ h16[b]^T  (NT GEMM); also writes eT = e^T so the
// softmax can read esym = e + e^T with two coalesced row streams.
__global__ __launch_bounds__(32)
void k_gemm_e(const _Float16* __restrict__ hA, const _Float16* __restrict__ h,
              float* __restrict__ e, float* __restrict__ eT) {
    int lane = threadIdx.x;
    int mb = blockIdx.x << 5, nb = blockIdx.y << 5, b = blockIdx.z;
    const _Float16* Ab = hA + (size_t)b * N * Dp;
    const _Float16* Bb = h  + (size_t)b * N * Dp;
    v8f acc[2][2] = {};
#pragma unroll
    for (int kb = 0; kb < Dp; kb += 32) {
        v16h a0 = load_a_frag(Ab, Dp, mb,      kb, lane);
        v16h a1 = load_a_frag(Ab, Dp, mb + 16, kb, lane);
        v16h b0 = load_bt_frag(Bb, Dp, nb,      kb, lane);
        v16h b1 = load_bt_frag(Bb, Dp, nb + 16, kb, lane);
        acc[0][0] = wmma_f16(a0, b0, acc[0][0]);
        acc[0][1] = wmma_f16(a0, b1, acc[0][1]);
        acc[1][0] = wmma_f16(a1, b0, acc[1][0]);
        acc[1][1] = wmma_f16(a1, b1, acc[1][1]);
    }
    int half = lane >> 4;
    float* eb  = e  + (size_t)b * N * N;
    float* ebT = eT + (size_t)b * N * N;
#pragma unroll
    for (int i = 0; i < 2; ++i) {
#pragma unroll
        for (int j = 0; j < 2; ++j) {
            int col  = nb + j * 16 + (lane & 15);
            int rowb = mb + i * 16 + half * 8;
#pragma unroll
            for (int r = 0; r < 8; ++r)
                eb[(size_t)(rowb + r) * N + col] = acc[i][j][r];
            *(v8f*)&ebT[(size_t)col * N + rowb] = acc[i][j];      // contiguous 32B
        }
    }
}

// out[b,N,D] = relu(att16[b] (NxN) @ h16[b] (NxDp)); B read from h16T (NT form)
__global__ __launch_bounds__(32)
void k_gemm_ah(const _Float16* __restrict__ att, const _Float16* __restrict__ hT,
               float* __restrict__ out) {
    int lane = threadIdx.x;
    int mb = blockIdx.x << 5, nb = blockIdx.y << 5, b = blockIdx.z;
    const _Float16* Ab = att + (size_t)b * N * N;
    const _Float16* Bb = hT  + (size_t)b * Dp * N;
    v8f acc[2][2] = {};
    for (int kb = 0; kb < N; kb += 32) {
        __builtin_prefetch(Ab + (size_t)(mb + (lane & 15)) * N + kb + 128, 0, 0);
        v16h a0 = load_a_frag(Ab, N, mb,      kb, lane);
        v16h a1 = load_a_frag(Ab, N, mb + 16, kb, lane);
        v16h b0 = load_bt_frag(Bb, N, nb,      kb, lane);
        v16h b1 = load_bt_frag(Bb, N, nb + 16, kb, lane);
        acc[0][0] = wmma_f16(a0, b0, acc[0][0]);
        acc[0][1] = wmma_f16(a0, b1, acc[0][1]);
        acc[1][0] = wmma_f16(a1, b0, acc[1][0]);
        acc[1][1] = wmma_f16(a1, b1, acc[1][1]);
    }
    int half = lane >> 4;
    float* ob = out + (size_t)b * N * D;
#pragma unroll
    for (int i = 0; i < 2; ++i) {
#pragma unroll
        for (int j = 0; j < 2; ++j) {
            int col = nb + j * 16 + (lane & 15);
            if (col < D) {
                int rowb = mb + i * 16 + half * 8;
#pragma unroll
                for (int r = 0; r < 8; ++r) {
                    float v = acc[i][j][r];
                    ob[(size_t)(rowb + r) * D + col] = v > 0.f ? v : 0.f;
                }
            }
        }
    }
}

// ---------------------------------------------------------------------------
// Fused symmetrize + mask + column softmax (axis=1) + *adj -> f16 att.
// MODE 0: adj read directly (c_adjs1). MODE 1: adj2 formulated on the fly
// from raw distances (saves a 38MB buffer + a full write/read pass).
// ---------------------------------------------------------------------------
template <int MODE>
__global__ void k_attn(const float* __restrict__ e, const float* __restrict__ eT,
                       const float* __restrict__ adjsrc,
                       const int* __restrict__ rbuf, const int* __restrict__ natoms,
                       const float* __restrict__ mu, const float* __restrict__ dev,
                       _Float16* __restrict__ att) {
    int tx = threadIdx.x, ty = threadIdx.y;
    int j = blockIdx.x * 32 + tx;
    int b = blockIdx.y;
    const float* eb  = e      + (size_t)b * N * N;
    const float* ebT = eT     + (size_t)b * N * N;
    const float* ab  = adjsrc + (size_t)b * N * N;
    int   r = 0, n = 0;
    float muv = 0.f, devv = 1.f;
    if (MODE == 1) { r = rbuf[b]; n = natoms[b]; muv = mu[0]; devv = dev[0]; }

    auto adjval = [&](int i, float a) -> float {
        if (MODE == 0) return a;
        bool selA = (i < r) && (j >= r) && (j < n);
        bool selB = (j < r) && (i >= r) && (i < n);
        float dm  = a - muv;
        float upd = (a <= 10.f) ? __expf(-(dm * dm) / devv) : 0.f;
        return (selA || selB) ? upd : a;
    };

    __shared__ float red[8][32];

    float m = -3.4e38f;
    for (int i = ty; i < N; i += 8) {
        float av = adjval(i, ab[(size_t)i * N + j]);
        float v  = (av > 0.f) ? (eb[(size_t)i * N + j] + ebT[(size_t)i * N + j]) : -9e15f;
        m = fmaxf(m, v);
    }
    red[ty][tx] = m; __syncthreads();
    if (ty == 0) {
#pragma unroll
        for (int k = 1; k < 8; ++k) m = fmaxf(m, red[k][tx]);
        red[0][tx] = m;
    }
    __syncthreads();
    m = red[0][tx];
    __syncthreads();

    float s = 0.f;
    for (int i = ty; i < N; i += 8) {
        float av = adjval(i, ab[(size_t)i * N + j]);
        float v  = (av > 0.f) ? (eb[(size_t)i * N + j] + ebT[(size_t)i * N + j]) : -9e15f;
        s += __expf(v - m);
    }
    red[ty][tx] = s; __syncthreads();
    if (ty == 0) {
#pragma unroll
        for (int k = 1; k < 8; ++k) s += red[k][tx];
        red[0][tx] = s;
    }
    __syncthreads();
    s = red[0][tx];
    float inv = 1.f / s;

    _Float16* ob = att + (size_t)b * N * N;
    for (int i = ty; i < N; i += 8) {
        float av = adjval(i, ab[(size_t)i * N + j]);
        float v  = (av > 0.f) ? (eb[(size_t)i * N + j] + ebT[(size_t)i * N + j]) : -9e15f;
        ob[(size_t)i * N + j] = (_Float16)(__expf(v - m) * inv * av);
    }
}

// Gate: coeff = sigmoid([x, hp] @ gw + gb); hp <- coeff*x + (1-coeff)*hp
__global__ void k_gate(const float* __restrict__ x, float* __restrict__ hp,
                       const float* __restrict__ gw, const float* __restrict__ gb) {
    int row  = blockIdx.x * 4 + threadIdx.y;
    int lane = threadIdx.x;
    const float* xr = x  + (size_t)row * D;
    float*       hr = hp + (size_t)row * D;
    float s = 0.f;
    for (int c = lane; c < D; c += 32)
        s += xr[c] * gw[c] + hr[c] * gw[D + c];
#pragma unroll
    for (int off = 16; off; off >>= 1) s += __shfl_xor(s, off, 32);
    float coeff = 1.f / (1.f + __expf(-(s + gb[0])));
    for (int c = lane; c < D; c += 32)
        hr[c] = coeff * xr[c] + (1.f - coeff) * hr[c];
}

// x = g2 - g1, fused with padded-f16 emission for the next layer
__global__ void k_diff(const float* __restrict__ g1, const float* __restrict__ g2,
                       float* __restrict__ x, _Float16* __restrict__ x16) {
    int idx = blockIdx.x * blockDim.x + threadIdx.x;
    if (idx >= B * N * Dp) return;
    int d  = idx % Dp;
    int bn = idx / Dp;
    float v = 0.f;
    if (d < D) {
        v = g2[(size_t)bn * D + d] - g1[(size_t)bn * D + d];
        x[(size_t)bn * D + d] = v;
    }
    x16[idx] = (_Float16)v;
}

__global__ void k_pool(const float* __restrict__ x, const int* __restrict__ natoms,
                       float* __restrict__ pooled) {
    int b = blockIdx.x, d = threadIdx.x;
    if (d >= D) return;
    int n = natoms[b];
    float s = 0.f;
    for (int i = 0; i < n; ++i) s += x[((size_t)b * N + i) * D + d];
    pooled[b * D + d] = s;
}

__global__ void k_mlp(const float* __restrict__ pooled,
                      const float* __restrict__ w0, const float* __restrict__ b0,
                      const float* __restrict__ w1, const float* __restrict__ b1,
                      const float* __restrict__ w2, const float* __restrict__ b2,
                      const float* __restrict__ w3, const float* __restrict__ b3,
                      float* __restrict__ out) {
    __shared__ float sp[D];
    __shared__ float sh[DF];
    __shared__ float red[DF];
    int t = threadIdx.x;
    for (int b = 0; b < B; ++b) {
        for (int i = t; i < D; i += DF) sp[i] = pooled[b * D + i];
        __syncthreads();
        float v = b0[t];
        for (int d = 0; d < D; ++d) v = fmaf(sp[d], w0[d * DF + t], v);
        v = fmaxf(v, 0.f);
        __syncthreads(); sh[t] = v; __syncthreads();
        v = b1[t];
        for (int j = 0; j < DF; ++j) v = fmaf(sh[j], w1[j * DF + t], v);
        v = fmaxf(v, 0.f);
        __syncthreads(); sh[t] = v; __syncthreads();
        v = b2[t];
        for (int j = 0; j < DF; ++j) v = fmaf(sh[j], w2[j * DF + t], v);
        v = fmaxf(v, 0.f);
        red[t] = v * w3[t];
        __syncthreads();
        for (int k = 64; k; k >>= 1) { if (t < k) red[t] += red[t + k]; __syncthreads(); }
        if (t == 0) out[b] = 1.f / (1.f + __expf(-(red[0] + b3[0])));
        __syncthreads();
    }
}

// ---------------------------------------------------------------------------
// Launcher
// ---------------------------------------------------------------------------
extern "C" void kernel_launch(void* const* d_in, const int* in_sizes, int n_in,
                              void* d_out, int out_size, void* d_ws, size_t ws_size,
                              hipStream_t stream) {
    const float* c_hs     = (const float*)d_in[0];
    const float* c_adjs1  = (const float*)d_in[1];
    const float* c_adjs2  = (const float*)d_in[2];
    const int*   c_valid  = (const int*)d_in[3];
    const int*   natoms   = (const int*)d_in[4];
    const float* W_emb    = (const float*)d_in[5];
    const float* gat_W    = (const float*)d_in[6];
    const float* gat_Wb   = (const float*)d_in[7];
    const float* gat_A    = (const float*)d_in[8];
    const float* gat_gw   = (const float*)d_in[9];
    const float* gat_gb   = (const float*)d_in[10];
    const float* fc_w0    = (const float*)d_in[11];
    const float* fc_b0    = (const float*)d_in[12];
    const float* fc_w1    = (const float*)d_in[13];
    const float* fc_b1    = (const float*)d_in[14];
    const float* fc_w2    = (const float*)d_in[15];
    const float* fc_b2    = (const float*)d_in[16];
    const float* fc_w3    = (const float*)d_in[17];
    const float* fc_b3    = (const float*)d_in[18];
    const float* mu       = (const float*)d_in[19];
    const float* devp     = (const float*)d_in[20];
    float* outp = (float*)d_out;

    constexpr size_t SZ_E    = (size_t)B * N * N * 4;
    constexpr size_t SZ_ATT  = (size_t)B * N * N * 2;
    constexpr size_t SZ_X    = (size_t)B * N * D * 4;
    constexpr size_t SZ_H16  = (size_t)B * N * Dp * 2;
    constexpr size_t SZ_W16  = (size_t)Dp * Dp * 2;
    constexpr size_t SZ_MISC = 16384;
    constexpr size_t WS_NEED = 2 * SZ_E + SZ_ATT + 3 * SZ_X + 4 * SZ_H16
                             + 2 * SZ_W16 + SZ_MISC;
    if (ws_size < WS_NEED) return;

    uint8_t* p = (uint8_t*)d_ws;
    float*     ebuf  = (float*)p;     p += SZ_E;
    float*     eTbuf = (float*)p;     p += SZ_E;
    _Float16*  att16 = (_Float16*)p;  p += SZ_ATT;
    float*     xbuf  = (float*)p;     p += SZ_X;
    float*     g1    = (float*)p;     p += SZ_X;
    float*     g2    = (float*)p;     p += SZ_X;
    _Float16*  x16   = (_Float16*)p;  p += SZ_H16;
    _Float16*  h16   = (_Float16*)p;  p += SZ_H16;
    _Float16*  h16T  = (_Float16*)p;  p += SZ_H16;   // [b][Dp][N]
    _Float16*  hA16  = (_Float16*)p;  p += SZ_H16;
    _Float16*  Wt16  = (_Float16*)p;  p += SZ_W16;
    _Float16*  At16  = (_Float16*)p;  p += SZ_W16;
    int*       rbuf  = (int*)p;       p += 256;
    float*     pooled= (float*)p;

    k_embed<<<(B * N * Dp + 255) / 256, 256, 0, stream>>>(c_hs, W_emb, xbuf, x16);
    k_rcount<<<B, 256, 0, stream>>>(c_valid, rbuf);

    dim3 w32(32);
    for (int k = 0; k < L; ++k) {
        k_convw<<<(Dp * Dp + 255) / 256, 256, 0, stream>>>(
            gat_W + (size_t)k * D * D, gat_A + (size_t)k * D * D, Wt16, At16);

        // h = x@W + Wb (row-major + transposed); hA = h@A; e/eT = hA @ h^T
        k_gemm_f16<<<dim3(N / 32, Dp / 32, B), w32, 0, stream>>>(
            x16, Wt16, gat_Wb + (size_t)k * D, h16, h16T);
        k_gemm_f16<<<dim3(N / 32, Dp / 32, B), w32, 0, stream>>>(
            h16, At16, nullptr, hA16, nullptr);
        k_gemm_e<<<dim3(N / 32, N / 32, B), w32, 0, stream>>>(hA16, h16, ebuf, eTbuf);

        // branch 1: adj1 (intra-molecule adjacency)
        k_attn<0><<<dim3(N / 32, B), dim3(32, 8), 0, stream>>>(
            ebuf, eTbuf, c_adjs1, rbuf, natoms, mu, devp, att16);
        k_gemm_ah<<<dim3(N / 32, Dp / 32, B), w32, 0, stream>>>(att16, h16T, g1);
        k_gate<<<(B * N) / 4, dim3(32, 4), 0, stream>>>(
            xbuf, g1, gat_gw + (size_t)k * 2 * D, gat_gb + k);

        // branch 2: adj2 formulated on the fly from raw cross-distances
        k_attn<1><<<dim3(N / 32, B), dim3(32, 8), 0, stream>>>(
            ebuf, eTbuf, c_adjs2, rbuf, natoms, mu, devp, att16);
        k_gemm_ah<<<dim3(N / 32, Dp / 32, B), w32, 0, stream>>>(att16, h16T, g2);
        k_gate<<<(B * N) / 4, dim3(32, 4), 0, stream>>>(
            xbuf, g2, gat_gw + (size_t)k * 2 * D, gat_gb + k);

        k_diff<<<(B * N * Dp + 255) / 256, 256, 0, stream>>>(g1, g2, xbuf, x16);
    }

    k_pool<<<B, 160, 0, stream>>>(xbuf, natoms, pooled);
    k_mlp<<<1, DF, 0, stream>>>(pooled, fc_w0, fc_b0, fc_w1, fc_b1,
                                fc_w2, fc_b2, fc_w3, fc_b3, outp);
}
